// Attentive_Fusion_28793460752827
// MI455X (gfx1250) — matrix operand order repaired and checked by reading, not compiled, not run on hardware
//
#include <hip/hip_runtime.h>
#include <hip/hip_bf16.h>
#include <math.h>

// ---------------------------------------------------------------------------
// Problem constants (reference: B=8, S=2048, D=768)
// ---------------------------------------------------------------------------
constexpr int kB    = 8;
constexpr int kS    = 2048;
constexpr int kD    = 768;
constexpr int kRows = kB * kS;          // 16384 flattened rows for projections
constexpr int kSChunks = 4;             // S1 split for occupancy (grid.z)
constexpr int kOut  = kB * kS;          // 16384 outputs
constexpr float kEps = 1e-7f;

typedef __bf16 bf16_t;
typedef __attribute__((ext_vector_type(16))) __bf16 v16bf;
typedef __attribute__((ext_vector_type(8)))  __bf16 v8bf;
typedef __attribute__((ext_vector_type(8)))  float  v8f;

// LDS row strides padded by 8 bf16 (16B) so 16 consecutive rows map to
// distinct bank groups.
constexpr int kLdaProj = 40;      // 32 K-elems + 8 pad (projection A tile)
constexpr int kLdbProj = 40;      // 32 K-elems + 8 pad (projection B^T tile)
constexpr int kLdAttn  = kD + 8;  // 776 bf16 per row (q/k tiles, attention)

static __device__ __forceinline__ v16bf cat8(v8bf lo, v8bf hi) {
  return __builtin_shufflevector(lo, hi, 0,1,2,3,4,5,6,7,8,9,10,11,12,13,14,15);
}

// Hardware transcendental tanh (V_TANH_F32 on gfx1250); TRANS ops co-execute
// with XDL WMMA, hiding the elementwise tail behind the matrix pipe.
static __device__ __forceinline__ float fast_tanh(float x) {
#if __has_builtin(__builtin_amdgcn_tanhf)
  return __builtin_amdgcn_tanhf(x);
#elif __has_builtin(__builtin_amdgcn_tanh_f32)
  return __builtin_amdgcn_tanh_f32(x);
#else
  return tanhf(x);
#endif
}

// CDNA5 async memory->LDS copy, 16 bytes per lane (GLOBAL_LOAD_ASYNC_TO_LDS_B128,
// tracked by ASYNCcnt). LDS flat addresses carry the LDS byte offset in the low
// 32 bits, so truncation gives VDST.
static __device__ __forceinline__ void async_copy_b128(void* lds_dst,
                                                       const void* gsrc) {
  const uint32_t ldsa = (uint32_t)(uintptr_t)lds_dst;
  const uint64_t ga   = (uint64_t)(uintptr_t)gsrc;
  asm volatile("global_load_async_to_lds_b128 %0, %1, off"
               :: "v"(ldsa), "v"(ga) : "memory");
}

template <int N>
static __device__ __forceinline__ void wait_asynccnt() {
  asm volatile("s_wait_asynccnt %0" :: "n"(N) : "memory");
}

// ---------------------------------------------------------------------------
// Kernel 1: Y[r, n] = bf16( X[r,:] @ W[:,n] + bias[n] )
//   X: [kRows, kD] f32    W: [kD, kD] f32    Y: [kRows, kD] bf16
// Block: 256 threads = 8 waves; tile 128(M) x 128(N); K stepped by 32.
// Wave w: wm = w&3 -> rows wm*32..+31 ; wn = w>>2 -> cols wn*64..+63
// => 2x4 = 8 independent v_wmma_f32_16x16x32_bf16 chains.
// (Synchronous staging kept: it must convert f32->bf16 in flight, which the
//  raw async byte-copies cannot do.)
// ---------------------------------------------------------------------------
__global__ __launch_bounds__(256)
void proj_gemm_bf16_kernel(const float* __restrict__ X,
                           const float* __restrict__ W,
                           const float* __restrict__ bias,
                           bf16_t* __restrict__ Y) {
  __shared__ bf16_t At[128 * kLdaProj];   // A tile, row-major [row][k]
  __shared__ bf16_t Bt[128 * kLdbProj];   // B tile, TRANSPOSED [col][k]

  const int tid  = threadIdx.x;
  const int lane = tid & 31;
  const int wave = tid >> 5;
  const int ln   = lane & 15;   // N column / M row within 16
  const int lh   = lane >> 4;   // half-wave select

  const int row0 = blockIdx.x * 128;
  const int col0 = blockIdx.y * 128;
  const int wm   = wave & 3;
  const int wn   = wave >> 2;

  // Accumulators initialized with bias (bias depends on N = column only).
  v8f acc[2][4];
  #pragma unroll
  for (int j = 0; j < 4; ++j) {
    const float bv = bias[col0 + wn * 64 + j * 16 + ln];
    #pragma unroll
    for (int i = 0; i < 2; ++i) {
      #pragma unroll
      for (int r = 0; r < 8; ++r) acc[i][j][r] = bv;
    }
  }

  for (int ks = 0; ks < kD / 32; ++ks) {
    const int k0 = ks * 32;
    __syncthreads();  // protect LDS reads of previous iteration

    // Stage A: 128x32 f32 -> bf16 LDS. 4096 elems, 16 per thread (4 x float4).
    #pragma unroll
    for (int i = 0; i < 4; ++i) {
      const int idx = tid + i * 256;
      const int r   = idx >> 3;
      const int c4  = (idx & 7) * 4;
      const float4 v = *(const float4*)(X + (size_t)(row0 + r) * kD + k0 + c4);
      bf16_t* dst = At + r * kLdaProj + c4;
      dst[0] = (bf16_t)v.x; dst[1] = (bf16_t)v.y;
      dst[2] = (bf16_t)v.z; dst[3] = (bf16_t)v.w;
    }
    // Stage B transposed: W[k0+K][col0+n] -> Bt[n][K].
    #pragma unroll
    for (int i = 0; i < 4; ++i) {
      const int idx = tid + i * 256;
      const int K   = idx >> 5;               // 0..31
      const int n4  = (idx & 31) * 4;
      const float4 v = *(const float4*)(W + (size_t)(k0 + K) * kD + col0 + n4);
      Bt[(n4 + 0) * kLdbProj + K] = (bf16_t)v.x;
      Bt[(n4 + 1) * kLdbProj + K] = (bf16_t)v.y;
      Bt[(n4 + 2) * kLdbProj + K] = (bf16_t)v.z;
      Bt[(n4 + 3) * kLdbProj + K] = (bf16_t)v.w;
    }
    __syncthreads();

    // A fragments (ISA 16-bit A 16x32 layout): lane holds row M=ln,
    // K chunks [lh*8, lh*8+8) and [lh*8+16, lh*8+24).
    v16bf afrag[2];
    #pragma unroll
    for (int i = 0; i < 2; ++i) {
      const bf16_t* p = At + (wm * 32 + i * 16 + ln) * kLdaProj + lh * 8;
      afrag[i] = cat8(*(const v8bf*)p, *(const v8bf*)(p + 16));
    }
    // B fragments: lane = column N=ln, K chunks [lh*8, lh*8+8) and [lh*8+16, lh*8+24).
    #pragma unroll
    for (int j = 0; j < 4; ++j) {
      const bf16_t* p = Bt + (wn * 64 + j * 16 + ln) * kLdbProj + lh * 8;
      const v16bf bfrag = cat8(*(const v8bf*)p, *(const v8bf*)(p + 16));
      #pragma unroll
      for (int i = 0; i < 2; ++i) {
        acc[i][j] = __builtin_amdgcn_wmma_f32_16x16x32_bf16(
            false, afrag[i], false, bfrag, (short)0, acc[i][j], false, false);
      }
    }
  }

  // Epilogue: C layout -> VGPR r holds M = r + lh*8, N = ln.
  #pragma unroll
  for (int i = 0; i < 2; ++i) {
    #pragma unroll
    for (int j = 0; j < 4; ++j) {
      const int col   = col0 + wn * 64 + j * 16 + ln;
      const int rbase = row0 + wm * 32 + i * 16 + lh * 8;
      #pragma unroll
      for (int r = 0; r < 8; ++r)
        Y[(size_t)(rbase + r) * kD + col] = (bf16_t)acc[i][j][r];
    }
  }
}

// ---------------------------------------------------------------------------
// Kernel 2: fused qk GEMM + exp(tanh) partial column reduction.
//   partial[zc, b, t] = { sum_s e*qk , sum_s e },  e = exp(tanh(qk[s,t]))
// Grid: (16 col-blocks, 8 batches, kSChunks s-chunks) = 512 workgroups so the
// LDS-bound (1 WG per WGP) kernel fills the whole device.
// Block: 256 threads / 8 waves, 128 output columns (16 per wave).
// LDS: persistent k slice [128][kD] (~194KB) + DOUBLE-BUFFERED q tiles
// [2][32][kD] (~97KB), both filled with GLOBAL_LOAD_ASYNC_TO_LDS_B128 so the
// next q tile streams in while 48 WMMAs run on the current one.
// ---------------------------------------------------------------------------
__global__ __launch_bounds__(256)
void attn_colreduce_kernel(const bf16_t* __restrict__ Q,
                           const bf16_t* __restrict__ Kmat,
                           float2* __restrict__ partials) {
  extern __shared__ char smem[];
  bf16_t* kT  = (bf16_t*)smem;                 // [128][kLdAttn]
  bf16_t* qT0 = kT + 128 * kLdAttn;            // [32][kLdAttn] buffer 0
  bf16_t* qT1 = qT0 + 32 * kLdAttn;            // [32][kLdAttn] buffer 1

  const int tid  = threadIdx.x;
  const int lane = tid & 31;
  const int wave = tid >> 5;
  const int ln   = lane & 15;
  const int lh   = lane >> 4;

  const int b  = blockIdx.y;
  const int t0 = blockIdx.x * 128;
  const int zc = blockIdx.z;                    // s-chunk index
  const bf16_t* Qb = Q    + (size_t)b * kS * kD;
  const bf16_t* Kb = Kmat + (size_t)b * kS * kD;

  constexpr int kNSt  = kS / (32 * kSChunks);   // 16 s-steps of 32 rows
  const int stile0 = zc * kNSt;                 // first 32-row tile of chunk

  // Async-stage persistent k slice: 128 rows x 768 bf16 = 12288 x 16B chunks,
  // 48 per thread.
  #pragma unroll 4
  for (int i = 0; i < 48; ++i) {
    const int idx = tid + i * 256;
    const int r = idx / 96;
    const int c = (idx % 96) * 8;
    async_copy_b128(kT + r * kLdAttn + c, Kb + (size_t)(t0 + r) * kD + c);
  }

  // Async-stage one q tile (32 rows x 768 bf16, 12 x 16B chunks per thread).
  auto stage_q = [&](bf16_t* buf, int stile) {
    #pragma unroll
    for (int i = 0; i < 12; ++i) {
      const int idx = tid + i * 256;            // 3072 chunks
      const int r = idx / 96;                   // 0..31
      const int c = (idx % 96) * 8;
      async_copy_b128(buf + r * kLdAttn + c,
                      Qb + (size_t)(stile * 32 + r) * kD + c);
    }
  };
  stage_q(qT0, stile0);

  float num = 0.0f, den = 0.0f;

  for (int st = 0; st < kNSt; ++st) {
    bf16_t* cur = (st & 1) ? qT1 : qT0;
    bf16_t* nxt = (st & 1) ? qT0 : qT1;

    if (st + 1 < kNSt) {
      stage_q(nxt, stile0 + st + 1);  // overlaps with this step's WMMAs
      wait_asynccnt<12>();            // ASYNCcnt completes in order: all
                                      // copies older than these 12 are done.
    } else {
      wait_asynccnt<0>();
    }
    __syncthreads();          // everyone's staging of `cur` (and kT) visible

    v8f acc0 = {}, acc1 = {};
    #pragma unroll
    for (int ks = 0; ks < kD / 32; ++ks) {
      const int k0 = ks * 32;
      // A = q rows (M = s): lane row ln (+16 for the second M tile),
      // K chunks lh*8 and lh*8+16.
      const bf16_t* pa0 = cur + ln * kLdAttn + k0 + lh * 8;
      const bf16_t* pa1 = cur + (16 + ln) * kLdAttn + k0 + lh * 8;
      const v16bf a0 = cat8(*(const v8bf*)pa0, *(const v8bf*)(pa0 + 16));
      const v16bf a1 = cat8(*(const v8bf*)pa1, *(const v8bf*)(pa1 + 16));
      // B = k^T: lane column n -> k row (t0 + wave*16 + n), K chunks lh*8 and lh*8+16.
      const bf16_t* pb = kT + (wave * 16 + ln) * kLdAttn + k0 + lh * 8;
      const v16bf bb = cat8(*(const v8bf*)pb, *(const v8bf*)(pb + 16));
      acc0 = __builtin_amdgcn_wmma_f32_16x16x32_bf16(
          false, a0, false, bb, (short)0, acc0, false, false);
      acc1 = __builtin_amdgcn_wmma_f32_16x16x32_bf16(
          false, a1, false, bb, (short)0, acc1, false, false);
    }

    // Elementwise e = exp(tanh(qk)): v_tanh_f32 + v_exp_f32 (TRANS ops,
    // co-execute with the XDL WMMA pipe).
    #pragma unroll
    for (int r = 0; r < 8; ++r) {
      const float v0 = acc0[r];
      const float e0 = __expf(fast_tanh(v0));
      num += e0 * v0;
      den += e0;
      const float v1 = acc1[r];
      const float e1 = __expf(fast_tanh(v1));
      num += e1 * v1;
      den += e1;
    }
    __syncthreads();          // all reads of `cur` done before it's re-filled
  }

  // Lane L and L+16 hold complementary M halves of the same column N=ln.
  num += __shfl_xor(num, 16, 32);
  den += __shfl_xor(den, 16, 32);
  if (lh == 0) {
    const int t = t0 + wave * 16 + ln;
    partials[(size_t)zc * kOut + (size_t)b * kS + t] = make_float2(num, den);
  }
}

// ---------------------------------------------------------------------------
// Kernel 3: combine the kSChunks partials -> out = num / (den + eps).
// ---------------------------------------------------------------------------
__global__ __launch_bounds__(256)
void attn_finalize_kernel(const float2* __restrict__ partials,
                          float* __restrict__ out) {
  const int t = blockIdx.x * 256 + threadIdx.x;   // 0 .. kOut-1
  float num = 0.0f, den = 0.0f;
  #pragma unroll
  for (int zc = 0; zc < kSChunks; ++zc) {
    const float2 p = partials[(size_t)zc * kOut + t];
    num += p.x;
    den += p.y;
  }
  out[t] = num / (den + kEps);
}

// ---------------------------------------------------------------------------
// Host launcher.
// d_in order: x1, x2, Wq, bq, Wk, bk (per setup_inputs()).
// d_out: [8, 2048] f32.
// d_ws layout: q bf16 (25.2MB) | k bf16 (25.2MB) | partials f32x2 (512KB).
// ---------------------------------------------------------------------------
extern "C" void kernel_launch(void* const* d_in, const int* in_sizes, int n_in,
                              void* d_out, int out_size, void* d_ws, size_t ws_size,
                              hipStream_t stream) {
  (void)in_sizes; (void)n_in; (void)out_size; (void)ws_size;
  const float* x1 = (const float*)d_in[0];
  const float* x2 = (const float*)d_in[1];
  const float* Wq = (const float*)d_in[2];
  const float* bq = (const float*)d_in[3];
  const float* Wk = (const float*)d_in[4];
  const float* bk = (const float*)d_in[5];
  float* out = (float*)d_out;

  bf16_t* qws = (bf16_t*)d_ws;
  bf16_t* kws = qws + (size_t)kRows * kD;
  float2* partials = (float2*)(kws + (size_t)kRows * kD);

  const dim3 gProj(kRows / 128, kD / 128);   // 128 x 6 blocks
  proj_gemm_bf16_kernel<<<gProj, 256, 0, stream>>>(x1, Wq, bq, qws);
  proj_gemm_bf16_kernel<<<gProj, 256, 0, stream>>>(x2, Wk, bk, kws);

  const size_t shAttn =
      (size_t)(128 + 2 * 32) * kLdAttn * sizeof(bf16_t);  // ~291 KB
  hipFuncSetAttribute((const void*)attn_colreduce_kernel,
                      hipFuncAttributeMaxDynamicSharedMemorySize, (int)shAttn);
  attn_colreduce_kernel<<<dim3(kS / 128, kB, kSChunks), 256, shAttn, stream>>>(
      qws, kws, partials);

  attn_finalize_kernel<<<kOut / 256, 256, 0, stream>>>(partials, out);
}
